// GCG_4183298146527
// MI455X (gfx1250) — matrix operand order repaired and checked by reading, not compiled, not run on hardware
//
#include <hip/hip_runtime.h>
#include <hip/hip_bf16.h>
#include <math.h>

// ---------------------------------------------------------------------------
// Problem dims (from reference)
// ---------------------------------------------------------------------------
#define Bsz 256
#define Ssz 30
#define Csz 62
#define Lsz 169
#define L1o 106   // 169 - 64 + 1
#define L2o 75    // 106 - 32 + 1
#define L3o 62    // 75  - 14 + 1
#define Kd1 64
#define Kd2 32
#define Kd3 14
#define OGf 64
#define OHf 64

typedef __attribute__((ext_vector_type(16))) _Float16 v16h;
typedef __attribute__((ext_vector_type(8)))  float    v8f;
typedef __attribute__((ext_vector_type(2)))  _Float16 h2;

// ---------------------------------------------------------------------------
// LDS plan (bytes). Regions reused across phases; offsets kept 32B aligned.
//   XT   : f16 [62][170]  staged x_t                      21,080 -> 21,088
//   BUF1 : f16 [64][112]  y1 (dw1 out) / y2 (dw2 out)     14,336
//   BUF2 : f16 [64][112]  z1 (pw1 out) / z2 (pw2 out)     14,336
//   WREG : f16 [64][64]   padded pointwise weight          8,192
//   MISC : arow(62 f32)@+0 ; v(169 f32)@+256 ; bias64@+960 1,280
// ---------------------------------------------------------------------------
#define XT_STR    170
#define OFF_XT    0
#define OFF_BUF1  21088
#define OFF_BUF2  (21088 + 14336)
#define OFF_WREG  (21088 + 14336 + 14336)
#define OFF_MISC  (21088 + 14336 + 14336 + 8192)
#define SMEM_BYTES (OFF_MISC + 1280)

__device__ __forceinline__ float sigmoidf_(float x) {
    return 1.0f / (1.0f + expf(-x));
}

// ---------------------------------------------------------------------------
// Padded 64x64 (f16) x 64xN (f16) -> 64xN f32 GEMM, WMMA tiled, bias fused.
// A (weights) in WREG, row-major 64x64, pad rows/cols zeroed.
// B (activations) row-major [64][bstride] halfs, K = row index; pads zeroed.
// D written back as f16 unconditionally: pad cells of Z are never consumed
// (z feeds only depthwise stages that read rows<62, cols<valid), and A/bias
// pad rows are zero, so no predication is needed in the epilogue.
//
// Fragment layouts per CDNA5 ISA 7.12.2 (wave32):
//  A 16x32 f16: lanes 0-15 / 16-31 both hold M=lane&15; VGPR i holds K pair
//    k = (i/4)*16 + (i%4)*2 + 8*(lane>>4).
//  B 32x16 f16: lane holds column N = lane&15; half s holds k = (s/8)*16 + 8*(lane>>4) + s%8.
//  C/D 16x16 f32: VGPR r -> M = r + 8*(lane>>4), N = lane&15.
// ---------------------------------------------------------------------------
__device__ __forceinline__ void wmma_pointwise(
    const _Float16* __restrict__ W,    // LDS, 64x64
    const _Float16* __restrict__ Bm,   // LDS, 64 x bstride
    int bstride,
    _Float16* __restrict__ Z,          // LDS, 64 x zstride
    int zstride, int nTiles,
    const float* __restrict__ sbias,   // LDS, 64 floats, rows 62/63 = 0
    int lane, int wave)
{
    const int total = 4 * nTiles;      // M is always 4 tiles (64 rows)
    const int hf = lane >> 4;
    const int lm = lane & 15;
    for (int tile = wave; tile < total; tile += 8) {
        const int mt = tile / nTiles, nt = tile % nTiles;
        const int mbase = mt * 16, nbase = nt * 16;
        const int m = mbase + lm;
        v8f acc = {};
#pragma unroll
        for (int kc = 0; kc < 2; ++kc) {
            const int kk = kc * 32;
            v16h a;
#pragma unroll
            for (int i = 0; i < 8; ++i) {
                const int k = kk + ((i >> 2) << 4) + ((i & 3) << 1) + (hf << 3);
                h2 p = *(const h2*)(W + m * 64 + k);
                a[2 * i]     = p.x;
                a[2 * i + 1] = p.y;
            }
            v16h bf;
#pragma unroll
            for (int s = 0; s < 16; ++s) {
                const int k = kk + ((s >> 3) << 4) + (hf << 3) + (s & 7);
                bf[s] = Bm[k * bstride + nbase + lm];
            }
            acc = __builtin_amdgcn_wmma_f32_16x16x32_f16(
                false, a, false, bf, (short)0, acc, false, false);
        }
        const int col = nbase + lm;
        _Float16* zp = Z + (mbase + 8 * hf) * zstride + col;
#pragma unroll
        for (int r = 0; r < 8; ++r) {
            zp[r * zstride] = (_Float16)(acc[r] + sbias[mbase + 8 * hf + r]);
        }
    }
}

// ---------------------------------------------------------------------------
// Kernel 1: per (b,t) tile — fused dsconv chain + adaptive adjacency row 61 +
// GCN output row -> gf[t][b][0:64]. Grid (B, S), 256 threads (8 wave32).
// ---------------------------------------------------------------------------
__global__ __launch_bounds__(256)
void fused_spatial_kernel(
    const float* __restrict__ x,    const float* __restrict__ adj,
    const float* __restrict__ dw1w, const float* __restrict__ dw1b,
    const float* __restrict__ pw1w, const float* __restrict__ pw1b,
    const float* __restrict__ dw2w, const float* __restrict__ dw2b,
    const float* __restrict__ pw2w, const float* __restrict__ pw2b,
    const float* __restrict__ dw3w, const float* __restrict__ dw3b,
    const float* __restrict__ pw3w, const float* __restrict__ pw3b,
    const float* __restrict__ gcnw, const float* __restrict__ gcnb,
    float* __restrict__ gf)
{
    __shared__ __align__(32) char smem[SMEM_BYTES];
    _Float16* sXT = (_Float16*)(smem + OFF_XT);
    _Float16* sB1 = (_Float16*)(smem + OFF_BUF1);
    _Float16* sB2 = (_Float16*)(smem + OFF_BUF2);
    _Float16* sW  = (_Float16*)(smem + OFF_WREG);
    float* sArow  = (float*)(smem + OFF_MISC);
    float* sV     = (float*)(smem + OFF_MISC + 256);
    float* sBias  = (float*)(smem + OFF_MISC + 960);

    const int b = blockIdx.x, t = blockIdx.y;
    const int tid = threadIdx.x, lane = tid & 31, wave = tid >> 5;
    const float* xt = x + ((size_t)b * Ssz + t) * (Csz * Lsz);

    // Warm L2 for this block's x_t slab (lowers to global_prefetch_b8).
    for (int i = tid * 32; i < Csz * Lsz; i += 256 * 32)
        __builtin_prefetch(xt + i, 0, 1);

    // P0: stage x_t (f16), padded pw1 weights, padded pw1 bias
    for (int i = tid; i < 4096; i += 256) {
        const int c = i >> 6, j = i & 63;
        sW[i] = (_Float16)((c < Csz && j < Csz) ? pw1w[c * Csz + j] : 0.0f);
    }
    if (tid < 64) sBias[tid] = (tid < Csz) ? pw1b[tid] : 0.0f;
    for (int i = tid; i < Csz * Lsz; i += 256) {
        const int c = i / Lsz, l = i - c * Lsz;
        sXT[c * XT_STR + l] = (_Float16)xt[i];
    }
    __syncthreads();

    // P1: depthwise conv 1 (K=64) -> sB1 (64x112 f16, pads zeroed: B operand)
    for (int i = tid; i < 64 * 112; i += 256) {
        const int c = i / 112, j = i - c * 112;
        float acc = 0.0f;
        if (c < Csz && j < L1o) {
            acc = dw1b[c];
            const float*    wr = dw1w + c * Kd1;
            const _Float16* xr = sXT + c * XT_STR + j;
#pragma unroll 8
            for (int k = 0; k < Kd1; ++k) acc += wr[k] * (float)xr[k];
        }
        sB1[i] = (_Float16)acc;
    }
    __syncthreads();

    // P2: pointwise conv 1 via WMMA -> sB2 (z1, 64x112)
    wmma_pointwise(sW, sB1, 112, sB2, 112, 7, sBias, lane, wave);
    __syncthreads();

    // P3: depthwise conv 2 (K=32) -> sB1 reused as 64x80 (pads zeroed);
    //     reload W with pw2, sBias with pw2_b
    for (int i = tid; i < 64 * 80; i += 256) {
        const int c = i / 80, j = i - c * 80;
        float acc = 0.0f;
        if (c < Csz && j < L2o) {
            acc = dw2b[c];
            const float*    wr = dw2w + c * Kd2;
            const _Float16* zr = sB2 + c * 112 + j;
#pragma unroll 8
            for (int k = 0; k < Kd2; ++k) acc += wr[k] * (float)zr[k];
        }
        sB1[i] = (_Float16)acc;
    }
    for (int i = tid; i < 4096; i += 256) {
        const int c = i >> 6, j = i & 63;
        sW[i] = (_Float16)((c < Csz && j < Csz) ? pw2w[c * Csz + j] : 0.0f);
    }
    if (tid < 64) sBias[tid] = (tid < Csz) ? pw2b[tid] : 0.0f;
    __syncthreads();

    // P4: pointwise conv 2 via WMMA -> sB2 reused as z2 (64x80)
    wmma_pointwise(sW, sB1, 80, sB2, 80, 5, sBias, lane, wave);
    __syncthreads();

    // P5: dw3 + pw3 row 61 fused; adaptive adjacency row: arow[j]
    if (tid < L3o) {
        const int j = tid;
        float acc = pw3b[Csz - 1];
        for (int c = 0; c < Csz; ++c) {
            float y = dw3b[c];
            const float*    wr = dw3w + c * Kd3;
            const _Float16* zr = sB2 + c * 80 + j;
#pragma unroll
            for (int k = 0; k < Kd3; ++k) y += wr[k] * (float)zr[k];
            acc += pw3w[(Csz - 1) * Csz + c] * y;
        }
        sArow[j] = sigmoidf_(acc + adj[(Csz - 1) * Csz + j]);
    }
    __syncthreads();

    // P6a: v[l] = sum_j arow[j] * x_t[j][l]   (re-associated GCN, row 61 only)
    if (tid < Lsz) {
        float acc = 0.0f;
        for (int j = 0; j < Csz; ++j)
            acc += sArow[j] * (float)sXT[j * XT_STR + tid];
        sV[tid] = acc;
    }
    __syncthreads();

    // P6b: g[f] = gcn_b[f] + sum_l v[l] * gcn_w[l][f]  -> gf[t][b][f]
    if (tid < OGf) {
        float acc = gcnb[tid];
        for (int l = 0; l < Lsz; ++l) acc += sV[l] * gcnw[l * OGf + tid];
        gf[((size_t)t * Bsz + b) * OGf + tid] = acc;
    }
}

// ---------------------------------------------------------------------------
// Kernel 2: GRU scan over 30 steps for the single live row (c = 61) per batch.
// One block per batch (64 threads = 2 wave32). Weights staged f16 in LDS.
// ---------------------------------------------------------------------------
__global__ __launch_bounds__(64)
void gru_scan_kernel(
    const float* __restrict__ gf,
    const float* __restrict__ Wr, const float* __restrict__ Wrb,
    const float* __restrict__ Wu, const float* __restrict__ Wub,
    const float* __restrict__ Wc, const float* __restrict__ Wcb,
    const float* __restrict__ br, const float* __restrict__ bu,
    const float* __restrict__ bc,
    float* __restrict__ out)
{
    __shared__ _Float16 sWr[OHf * 128], sWu[OHf * 128], sWc[OHf * 128];
    __shared__ float sg[OHf], sh[OHf], srh[OHf];

    const int b = blockIdx.x, f = threadIdx.x;
    for (int i = f; i < OHf * 128; i += 64) {
        sWr[i] = (_Float16)Wr[i];
        sWu[i] = (_Float16)Wu[i];
        sWc[i] = (_Float16)Wc[i];
    }
    float h = 0.0f;
    sh[f] = 0.0f;
    const float biasR = Wrb[f] + br[f];
    const float biasU = Wub[f] + bu[f];
    const float biasC = Wcb[f] + bc[f];
    __syncthreads();

    for (int t = 0; t < Ssz; ++t) {
        sg[f] = gf[((size_t)t * Bsz + b) * OGf + f];
        __syncthreads();

        float aR = biasR, aU = biasU;
        const _Float16* wr = sWr + f * 128;
        const _Float16* wu = sWu + f * 128;
#pragma unroll 4
        for (int k = 0; k < OHf; ++k) {
            const float gk = sg[k], hk = sh[k];
            aR += (float)wr[k] * gk + (float)wr[64 + k] * hk;
            aU += (float)wu[k] * gk + (float)wu[64 + k] * hk;
        }
        const float r = sigmoidf_(aR);
        const float u = sigmoidf_(aU);
        srh[f] = r * h;
        __syncthreads();

        float aC = biasC;
        const _Float16* wc = sWc + f * 128;
#pragma unroll 4
        for (int k = 0; k < OHf; ++k)
            aC += (float)wc[k] * sg[k] + (float)wc[64 + k] * srh[k];
        h = u * h + (1.0f - u) * tanhf(aC);
        __syncthreads();
        sh[f] = h;
    }
    out[b * OHf + f] = h;
}

// ---------------------------------------------------------------------------
extern "C" void kernel_launch(void* const* d_in, const int* in_sizes, int n_in,
                              void* d_out, int out_size, void* d_ws, size_t ws_size,
                              hipStream_t stream) {
    const float* x    = (const float*)d_in[0];
    const float* adj  = (const float*)d_in[1];
    const float* dw1w = (const float*)d_in[2];
    const float* dw1b = (const float*)d_in[3];
    const float* pw1w = (const float*)d_in[4];
    const float* pw1b = (const float*)d_in[5];
    const float* dw2w = (const float*)d_in[6];
    const float* dw2b = (const float*)d_in[7];
    const float* pw2w = (const float*)d_in[8];
    const float* pw2b = (const float*)d_in[9];
    const float* dw3w = (const float*)d_in[10];
    const float* dw3b = (const float*)d_in[11];
    const float* pw3w = (const float*)d_in[12];
    const float* pw3b = (const float*)d_in[13];
    const float* gcnw = (const float*)d_in[14];
    const float* gcnb = (const float*)d_in[15];
    const float* Wr   = (const float*)d_in[16];
    const float* Wrb  = (const float*)d_in[17];
    const float* Wu   = (const float*)d_in[18];
    const float* Wub  = (const float*)d_in[19];
    const float* Wc   = (const float*)d_in[20];
    const float* Wcb  = (const float*)d_in[21];
    const float* br   = (const float*)d_in[22];
    const float* bu   = (const float*)d_in[23];
    const float* bc   = (const float*)d_in[24];

    float* gf = (float*)d_ws;   // (S, B, 64) f32 = 1.97 MB

    dim3 grid1(Bsz, Ssz);
    fused_spatial_kernel<<<grid1, 256, 0, stream>>>(
        x, adj, dw1w, dw1b, pw1w, pw1b, dw2w, dw2b, pw2w, pw2b,
        dw3w, dw3b, pw3w, pw3b, gcnw, gcnb, gf);

    gru_scan_kernel<<<Bsz, 64, 0, stream>>>(
        gf, Wr, Wrb, Wu, Wub, Wc, Wcb, br, bu, bc, (float*)d_out);
}
